// sparse_3D_unet_80607946212026
// MI455X (gfx1250) — matrix operand order repaired and checked by reading, not compiled, not run on hardware
//
#include <hip/hip_runtime.h>
#include <hip/hip_bf16.h>
#include <stdint.h>

// ---------------------------------------------------------------------------
// Sparse 3D U-Net forward for MI455X (gfx1250, wave32, WMMA).
// Gather-bound workload: f16 activations padded to 32-channel stride so A
// fragments are unconditional b128 loads; weights pre-packed into exact
// B-fragment order; one wave computes a 16-point tile x all out-channel tiles
// (up to 3 WMMA accumulators) so the gathered A is reused across octiles.
// ---------------------------------------------------------------------------

typedef __attribute__((ext_vector_type(16))) _Float16 v16h;
typedef __attribute__((ext_vector_type(8)))  _Float16 v8h;
typedef __attribute__((ext_vector_type(8)))  float    v8f;

#define TPB 256

// ------------------------------ elementwise --------------------------------

__global__ void k_zero_f32(float* p, long n) {
  long i = (long)blockIdx.x * blockDim.x + threadIdx.x;
  if (i < n) p[i] = 0.f;
}

// f32 compact (N,C) -> f16 padded (N+1, Cpad), zero padding + pad row
__global__ void k_cvt_padp(const float* __restrict__ x, _Float16* __restrict__ y,
                           long N, int C, int Cpad) {
  long i = (long)blockIdx.x * blockDim.x + threadIdx.x;
  long tot = (N + 1) * (long)Cpad;
  if (i >= tot) return;
  int c = (int)(i % Cpad);
  long row = i / Cpad;
  y[i] = (row < N && c < C) ? (_Float16)x[row * C + c] : (_Float16)0.f;
}

__global__ void k_add_inplace(float* __restrict__ x, const float* __restrict__ y, long n) {
  long i = (long)blockIdx.x * blockDim.x + threadIdx.x;
  if (i < n) x[i] += y[i];
}

__global__ void k_concat2(const float* __restrict__ x, const float* __restrict__ u,
                          float* __restrict__ c, long N, int C) {
  long i = (long)blockIdx.x * blockDim.x + threadIdx.x;
  long total = N * (long)(2 * C);
  if (i >= total) return;
  int cc = (int)(i % (2 * C));
  long n = i / (2 * C);
  c[i] = (cc < C) ? x[n * C + cc] : u[n * C + (cc - C)];
}

__global__ void k_bias_add(float* __restrict__ o, const float* __restrict__ b, long N, int C) {
  long i = (long)blockIdx.x * blockDim.x + threadIdx.x;
  if (i < N * (long)C) o[i] += b[i % C];
}

// ------------------------------ batch norm ---------------------------------

// per-channel sum / sumsq, LDS partials + global float atomics
__global__ void k_bn_reduce(const float* __restrict__ X, long N, int C,
                            float* __restrict__ sums) {
  __shared__ float ls[128];  // 2*C <= 96
  for (int i = threadIdx.x; i < 2 * C; i += blockDim.x) ls[i] = 0.f;
  __syncthreads();
  long total = N * (long)C;
  for (long i = (long)blockIdx.x * blockDim.x + threadIdx.x; i < total;
       i += (long)gridDim.x * blockDim.x) {
    float v = X[i];
    int c = (int)(i % C);
    atomicAdd(&ls[c], v);
    atomicAdd(&ls[C + c], v * v);
  }
  __syncthreads();
  for (int i = threadIdx.x; i < 2 * C; i += blockDim.x) atomicAdd(&sums[i], ls[i]);
}

// y = relu((x-mu)*rsqrt(var+eps)*g + b) as f16 into padded (N+1, Cpad) layout
__global__ void k_bn_apply_relu_f16p(const float* __restrict__ X, const float* __restrict__ sums,
                                     const float* __restrict__ g, const float* __restrict__ b,
                                     _Float16* __restrict__ Y, long N, int C, int Cpad) {
  long i = (long)blockIdx.x * blockDim.x + threadIdx.x;
  long tot = (N + 1) * (long)Cpad;
  if (i >= tot) return;
  int c = (int)(i % Cpad);
  long row = i / Cpad;
  if (row >= N || c >= C) { Y[i] = (_Float16)0.f; return; }
  float invN = 1.f / (float)N;
  float mu = sums[c] * invN;
  float var = sums[C + c] * invN - mu * mu;
  float v = (X[row * C + c] - mu) * rsqrtf(var + 1e-4f) * g[c] + b[c];
  Y[i] = (_Float16)fmaxf(v, 0.f);
}

// ------------------------------ up scatter ---------------------------------

// contrib[n,k,:] = y[n,:] @ W[k]  scattered-add into fine rows (drop idx>=Nf)
// Y is f16 padded with row stride Cbpad.
__global__ void k_up_scatter(const _Float16* __restrict__ Y, const float* __restrict__ W,
                             const int* __restrict__ rule, float* __restrict__ out,
                             long Nc, long Nf, int Cb, int Cbpad, int Ca) {
  long t = (long)blockIdx.x * blockDim.x + threadIdx.x;
  if (t >= Nc * 8) return;
  long n = t >> 3;
  int k = (int)(t & 7);
  int idx = rule[n * 8 + k];
  if (idx < 0 || (long)idx >= Nf) return;
  for (int a = 0; a < Ca; ++a) {
    float s = 0.f;
    for (int bb = 0; bb < Cb; ++bb)
      s += (float)Y[n * Cbpad + bb] * W[((long)k * Cb + bb) * Ca + a];
    atomicAdd(&out[(long)idx * Ca + a], s);
  }
}

// --------------------------- weight B-packing ------------------------------
// Pack W (R, Cin, Cout) f32 into f16 B fragments:
//   P[((r*kch + kc)*oct + o)*512 + lane*16 + e]
//   lane: column = o*16 + (lane&15); K = kc*32 + ((lane&16)?16:0) + e
__global__ void k_pack_w(const float* __restrict__ W, _Float16* __restrict__ P,
                         int R, int Cin, int Cout) {
  int kch = (Cin + 31) >> 5;
  int oct = (Cout + 15) >> 4;
  long total = (long)R * kch * oct * 512;
  long i = (long)blockIdx.x * blockDim.x + threadIdx.x;
  if (i >= total) return;
  int e = (int)(i & 15);
  int lane = (int)((i >> 4) & 31);
  long frag = i >> 9;
  int o = (int)(frag % oct);
  long t = frag / oct;
  int kc = (int)(t % kch);
  int r = (int)(t / kch);
  int col = o * 16 + (lane & 15);
  int K = kc * 32 + ((lane & 16) ? 16 : 0) + e;
  P[i] = (K < Cin && col < Cout) ? (_Float16)W[((long)r * Cin + K) * Cout + col]
                                 : (_Float16)0.f;
}

// --------------------------- gather-GEMM (WMMA) ----------------------------
// Out[n, :] = sum_r X[rules[n,r], :] @ W[r]   (rules==nullptr -> identity, R=1)
// X: (Nsrc+1, Cpad) f16, zero-padded rows/channels. Wp: packed B fragments.
// One wave: 16-point tile x ALL out-channel tiles (<=3), reusing gathered A.
__global__ void k_gather_gemm_wmma(const _Float16* __restrict__ X, const int* __restrict__ rules,
                                   const _Float16* __restrict__ Wp, float* __restrict__ Out,
                                   int Ndst, int Nsrc, int Cpad, int Cout, int R, int nTiles) {
  int tile = blockIdx.x * (blockDim.x >> 5) + (threadIdx.x >> 5);
  if (tile >= nTiles) return;  // wave-uniform: EXEC stays all-1s for WMMA
  int lane = threadIdx.x & 31;
  int rlo = lane & 15;
  bool hi = (lane & 16) != 0;
  int n = tile * 16 + rlo;
  int kchunks = Cpad >> 5;
  int octiles = (Cout + 15) >> 4;

  v8f acc0 = {}, acc1 = {}, acc2 = {};

  for (int r = 0; r < R; ++r) {
    int idx = (n < Ndst) ? (rules ? rules[(long)n * R + r] : n) : Nsrc;
    const _Float16* xrow = X + (long)idx * Cpad;
    if (rules && r + 1 < R) {  // CDNA5 global_prefetch_b8 on next gathered row
      int idx2 = (n < Ndst) ? rules[(long)n * R + r + 1] : Nsrc;
      __builtin_prefetch(X + (long)idx2 * Cpad, 0, 1);
    }
    const _Float16* wr = Wp + ((((long)r * kchunks) * octiles) << 9) + (lane << 4);
    for (int kc = 0; kc < kchunks; ++kc) {
      // A fragment: two contiguous aligned b128 loads per lane
      const _Float16* xk = xrow + kc * 32 + (hi ? 16 : 0);
      v8h a0 = *(const v8h*)(xk);
      v8h a1 = *(const v8h*)(xk + 8);
      v16h a = __builtin_shufflevector(a0, a1, 0, 1, 2, 3, 4, 5, 6, 7,
                                       8, 9, 10, 11, 12, 13, 14, 15);
      const _Float16* wkc = wr + (((long)kc * octiles) << 9);
      {
        v8h b0 = *(const v8h*)(wkc);
        v8h b1 = *(const v8h*)(wkc + 8);
        v16h b = __builtin_shufflevector(b0, b1, 0, 1, 2, 3, 4, 5, 6, 7,
                                         8, 9, 10, 11, 12, 13, 14, 15);
        acc0 = __builtin_amdgcn_wmma_f32_16x16x32_f16(false, a, false, b,
                                                      (short)0, acc0, false, false);
      }
      if (octiles > 1) {
        v8h b0 = *(const v8h*)(wkc + 512);
        v8h b1 = *(const v8h*)(wkc + 520);
        v16h b = __builtin_shufflevector(b0, b1, 0, 1, 2, 3, 4, 5, 6, 7,
                                         8, 9, 10, 11, 12, 13, 14, 15);
        acc1 = __builtin_amdgcn_wmma_f32_16x16x32_f16(false, a, false, b,
                                                      (short)0, acc1, false, false);
      }
      if (octiles > 2) {
        v8h b0 = *(const v8h*)(wkc + 1024);
        v8h b1 = *(const v8h*)(wkc + 1032);
        v16h b = __builtin_shufflevector(b0, b1, 0, 1, 2, 3, 4, 5, 6, 7,
                                         8, 9, 10, 11, 12, 13, 14, 15);
        acc2 = __builtin_amdgcn_wmma_f32_16x16x32_f16(false, a, false, b,
                                                      (short)0, acc2, false, false);
      }
    }
  }

  // D layout: VGPR j -> row (hi?8:0)+j, col = lane&15 (+16*oct)
#pragma unroll
  for (int j = 0; j < 8; ++j) {
    int m = tile * 16 + (hi ? 8 : 0) + j;
    if (m >= Ndst) continue;
    float* orow = Out + (long)m * Cout;
    if (rlo < Cout) orow[rlo] = acc0[j];
    if (octiles > 1 && 16 + rlo < Cout) orow[16 + rlo] = acc1[j];
    if (octiles > 2 && 32 + rlo < Cout) orow[32 + rlo] = acc2[j];
  }
}

// ------------------------------- host side ---------------------------------

extern "C" void kernel_launch(void* const* d_in, const int* in_sizes, int n_in,
                              void* d_out, int out_size, void* d_ws, size_t ws_size,
                              hipStream_t stream) {
  (void)n_in; (void)out_size; (void)ws_size;
  static const int pl[6] = {4, 8, 12, 16, 20, 24};
  auto cpad = [](int C) { return (C + 31) & ~31; };

  // ---- input mapping: jax tree-flatten order (dict keys sorted) ----
  struct LvlP {
    const float *down_b, *down_g, *down_w, *post_b1, *post_b2, *post_g1, *post_g2,
                *post_nin, *post_w1, *post_w2, *pre_b1, *pre_b2, *pre_g1, *pre_g2,
                *pre_w1, *pre_w2, *up_b, *up_g, *up_w;
  };
  const float* feats = (const float*)d_in[1];
  int p = 2;
  LvlP L[6] = {};
  for (int l = 0; l < 5; ++l) {
    L[l].down_b  = (const float*)d_in[p++]; L[l].down_g  = (const float*)d_in[p++];
    L[l].down_w  = (const float*)d_in[p++];
    L[l].post_b1 = (const float*)d_in[p++]; L[l].post_b2 = (const float*)d_in[p++];
    L[l].post_g1 = (const float*)d_in[p++]; L[l].post_g2 = (const float*)d_in[p++];
    L[l].post_nin= (const float*)d_in[p++];
    L[l].post_w1 = (const float*)d_in[p++]; L[l].post_w2 = (const float*)d_in[p++];
    L[l].pre_b1  = (const float*)d_in[p++]; L[l].pre_b2  = (const float*)d_in[p++];
    L[l].pre_g1  = (const float*)d_in[p++]; L[l].pre_g2  = (const float*)d_in[p++];
    L[l].pre_w1  = (const float*)d_in[p++]; L[l].pre_w2  = (const float*)d_in[p++];
    L[l].up_b    = (const float*)d_in[p++]; L[l].up_g    = (const float*)d_in[p++];
    L[l].up_w    = (const float*)d_in[p++];
  }
  L[5].pre_b1 = (const float*)d_in[p++]; L[5].pre_b2 = (const float*)d_in[p++];
  L[5].pre_g1 = (const float*)d_in[p++]; L[5].pre_g2 = (const float*)d_in[p++];
  L[5].pre_w1 = (const float*)d_in[p++]; L[5].pre_w2 = (const float*)d_in[p++];
  const float* lin_b = (const float*)d_in[p++];
  const float* lin_w = (const float*)d_in[p++];
  const float* out_b = (const float*)d_in[p++];
  const float* out_g = (const float*)d_in[p++];
  const float* w_in  = (const float*)d_in[p++];
  const int* down_r[5]; for (int l = 0; l < 5; ++l) down_r[l] = (const int*)d_in[p++];
  const int* nbr_r[6];
  int nbr_base = p;
  for (int l = 0; l < 6; ++l) nbr_r[l] = (const int*)d_in[p++];

  long N[6];
  for (int l = 0; l < 6; ++l) N[l] = (long)in_sizes[nbr_base + l] / 27;

  auto cdiv = [](long a, long b) { return (a + b - 1) / b; };
  auto el = [&](long n) { return dim3((unsigned)((n + TPB - 1) / TPB)); };

  // ---- workspace bump allocator ----
  char* wp = (char*)d_ws;
  auto alloc = [&](size_t bytes) -> void* {
    void* r = (void*)wp;
    wp += (bytes + 255) & ~(size_t)255;
    return r;
  };

  // ---- pack all weights into B-fragment f16 layout once ----
  auto packW = [&](const float* w, int R, int Cin, int Cout) -> _Float16* {
    long total = (long)R * ((Cin + 31) >> 5) * ((Cout + 15) >> 4) * 512;
    _Float16* d = (_Float16*)alloc(total * sizeof(_Float16));
    k_pack_w<<<el(total), dim3(TPB), 0, stream>>>(w, d, R, Cin, Cout);
    return d;
  };
  _Float16 *pre_w1_h[6], *pre_w2_h[6], *down_w_h[5], *post_w1_h[5], *post_w2_h[5], *nin_h[5];
  for (int l = 0; l < 6; ++l) {
    int a = pl[l];
    pre_w1_h[l] = packW(L[l].pre_w1, 27, a, a);
    pre_w2_h[l] = packW(L[l].pre_w2, 27, a, a);
    if (l < 5) {
      int b = pl[l + 1];
      down_w_h[l]  = packW(L[l].down_w, 8, a, b);
      post_w1_h[l] = packW(L[l].post_w1, 27, 2 * a, a);
      post_w2_h[l] = packW(L[l].post_w2, 27, a, a);
      nin_h[l]     = packW(L[l].post_nin, 1, 2 * a, a);
    }
  }
  _Float16* w_in_h = packW(w_in, 27, 3, 4);
  _Float16* linw_h = packW(lin_w, 1, 4, 48);

  // ---- persistent per-level activations + rotating scratch ----
  float* x[6];
  for (int l = 0; l < 6; ++l) x[l] = (float*)alloc((size_t)N[l] * pl[l] * sizeof(float));
  long maxF32 = 0, maxF16 = 0;
  for (int l = 0; l < 6; ++l) {
    long f32n = N[l] * 2L * pl[l];
    long f16n = (N[l] + 1) * (long)cpad(2 * pl[l]);
    if (f32n > maxF32) maxF32 = f32n;
    if (f16n > maxF16) maxF16 = f16n;
  }
  float* SCR0 = (float*)alloc((size_t)maxF32 * sizeof(float));
  float* SCR1 = (float*)alloc((size_t)maxF32 * sizeof(float));
  float* SCR2 = (float*)alloc((size_t)maxF32 * sizeof(float));
  _Float16* T16A = (_Float16*)alloc((size_t)maxF16 * sizeof(_Float16));
  _Float16* T16B = (_Float16*)alloc((size_t)maxF16 * sizeof(_Float16));
  float* bnsums = (float*)alloc(128 * sizeof(float));

  // ---- launch helpers ----
  auto bn = [&](const float* X, long n, int C, const float* g, const float* b, _Float16* Y) {
    k_zero_f32<<<dim3(1), dim3(128), 0, stream>>>(bnsums, 128);
    long tot = n * (long)C;
    long rb = cdiv(tot, TPB); if (rb > 1024) rb = 1024;
    k_bn_reduce<<<dim3((unsigned)rb), dim3(TPB), 0, stream>>>(X, n, C, bnsums);
    int Cp = cpad(C);
    k_bn_apply_relu_f16p<<<el((n + 1) * Cp), dim3(TPB), 0, stream>>>(
        X, bnsums, g, b, Y, n, C, Cp);
  };
  auto gg = [&](const _Float16* X, const int* rules, const _Float16* Wp, float* Out,
                long Ndst, long Nsrc, int Cin, int Cout, int R) {
    int tiles = (int)((Ndst + 15) >> 4);
    k_gather_gemm_wmma<<<dim3((unsigned)cdiv(tiles, 8)), dim3(256), 0, stream>>>(
        X, rules, Wp, Out, (int)Ndst, (int)Nsrc, cpad(Cin), Cout, R, tiles);
  };

  // ================================ forward ================================
  // input conv: feats (N0,3) -> x0 (N0,4)
  k_cvt_padp<<<el((N[0] + 1) * cpad(3)), dim3(TPB), 0, stream>>>(feats, T16B, N[0], 3, cpad(3));
  gg(T16B, nbr_r[0], w_in_h, x[0], N[0], N[0], 3, 4, 27);

  // ---- descend: pre residual block per level, then down conv ----
  for (int l = 0; l < 6; ++l) {
    int a = pl[l];
    bn(x[l], N[l], a, L[l].pre_g1, L[l].pre_b1, T16A);
    gg(T16A, nbr_r[l], pre_w1_h[l], SCR0, N[l], N[l], a, a, 27);
    bn(SCR0, N[l], a, L[l].pre_g2, L[l].pre_b2, T16A);
    gg(T16A, nbr_r[l], pre_w2_h[l], SCR1, N[l], N[l], a, a, 27);
    k_add_inplace<<<el(N[l] * a), dim3(TPB), 0, stream>>>(x[l], SCR1, N[l] * a);
    if (l < 5) {
      int b = pl[l + 1];
      bn(x[l], N[l], a, L[l].down_g, L[l].down_b, T16A);
      gg(T16A, down_r[l], down_w_h[l], x[l + 1], N[l + 1], N[l], a, b, 8);
    }
  }

  // ---- ascend: up conv, concat, post residual block with nin shortcut ----
  for (int l = 4; l >= 0; --l) {
    int a = pl[l], b = pl[l + 1];
    bn(x[l + 1], N[l + 1], b, L[l].up_g, L[l].up_b, T16A);
    k_zero_f32<<<el(N[l] * a), dim3(TPB), 0, stream>>>(SCR0, N[l] * a);
    k_up_scatter<<<el(N[l + 1] * 8), dim3(TPB), 0, stream>>>(
        T16A, L[l].up_w, down_r[l], SCR0, N[l + 1], N[l], b, cpad(b), a);
    k_concat2<<<el(N[l] * 2 * a), dim3(TPB), 0, stream>>>(x[l], SCR0, SCR1, N[l], a);
    // y = subm(bn_relu(cat)) ; y = subm(bn_relu(y)) ; x = cat@nin + y
    bn(SCR1, N[l], 2 * a, L[l].post_g1, L[l].post_b1, T16A);
    gg(T16A, nbr_r[l], post_w1_h[l], SCR2, N[l], N[l], 2 * a, a, 27);
    bn(SCR2, N[l], a, L[l].post_g2, L[l].post_b2, T16A);
    gg(T16A, nbr_r[l], post_w2_h[l], SCR0, N[l], N[l], a, a, 27);
    k_cvt_padp<<<el((N[l] + 1) * cpad(2 * a)), dim3(TPB), 0, stream>>>(
        SCR1, T16B, N[l], 2 * a, cpad(2 * a));
    gg(T16B, nullptr, nin_h[l], x[l], N[l], N[l], 2 * a, a, 1);
    k_add_inplace<<<el(N[l] * a), dim3(TPB), 0, stream>>>(x[l], SCR0, N[l] * a);
  }

  // ---- head: bn_relu -> linear (4 -> 48) + bias ----
  bn(x[0], N[0], 4, out_g, out_b, T16A);
  gg(T16A, nullptr, linw_h, (float*)d_out, N[0], N[0], 4, 48, 1);
  k_bias_add<<<el(N[0] * 48), dim3(TPB), 0, stream>>>((float*)d_out, lin_b, N[0], 48);
}
